// MultiHeadSelfAttention_71932112273744
// MI455X (gfx1250) — matrix operand order repaired and checked by reading, not compiled, not run on hardware
//
#include <hip/hip_runtime.h>

// ---------- types ----------
typedef __attribute__((ext_vector_type(16))) __bf16 bf16x16;
typedef __attribute__((ext_vector_type(8)))  float  f32x8;
typedef __attribute__((ext_vector_type(4)))  unsigned int u32x4;
typedef __attribute__((ext_vector_type(8)))  int i32x8;
typedef __attribute__((ext_vector_type(4)))  int i32x4;

union FragB { unsigned int u[8]; bf16x16 v; };

__device__ inline f32x8 wmma_bf16(const FragB& a, const FragB& b, f32x8 c) {
  return __builtin_amdgcn_wmma_f32_16x16x32_bf16(
      /*neg_a=*/false, a.v, /*neg_b=*/false, b.v,
      /*c_mod=*/(short)0, c, /*reuse_a=*/false, /*reuse_b=*/false);
}

__device__ inline unsigned short f2bf(float f) {
  unsigned u = __float_as_uint(f);
  unsigned r = u + 0x7FFFu + ((u >> 16) & 1u);   // round-to-nearest-even
  return (unsigned short)(r >> 16);
}

// A-fragment (16x32 bf16): u[0..3] = 8 shorts at p, u[4..7] = 8 shorts at p+8
__device__ inline void ldfragA(FragB& f, const unsigned short* p) {
  const uint4 lo = *(const uint4*)(p);
  const uint4 hi = *(const uint4*)(p + 8);
  f.u[0] = lo.x; f.u[1] = lo.y; f.u[2] = lo.z; f.u[3] = lo.w;
  f.u[4] = hi.x; f.u[5] = hi.y; f.u[6] = hi.z; f.u[7] = hi.w;
}
// B-fragment (32x16 bf16): 16 contiguous shorts at p
__device__ inline void ldfragB(FragB& f, const unsigned short* p) {
  const uint4 lo = *(const uint4*)(p);
  const uint4 hi = *(const uint4*)(p + 8);
  f.u[0] = lo.x; f.u[1] = lo.y; f.u[2] = lo.z; f.u[3] = lo.w;
  f.u[4] = hi.x; f.u[5] = hi.y; f.u[6] = hi.z; f.u[7] = hi.w;
}

// ---------- problem constants ----------
#define EMB   768
#define SEQ   2048
#define NB    4
#define NH    12
#define HD    64
#define BH    (NB*NH)          // 48
#define MTOT  (NB*SEQ)         // 8192
#define N_QKV (3*EMB)          // 2304
#define PANEL_SHORTS (64 * EMB)        // 49152 shorts = 96 KB

// ---------- TDM: stage a 64 x 768 bf16 weight panel into LDS ----------
// D# per CDNA5 ISA (2-D tensor, groups 2/3 zero).
// 6-arg builtin form (amdgpu-toolchain / therock-10.0).
__device__ inline void stage_panel_lds(unsigned short* bpanel,
                                       const unsigned short* gsrc, int tid) {
#if __has_builtin(__builtin_amdgcn_tensor_load_to_lds)
  if (tid < 32) {                       // one wave issues the DMA
    unsigned long long ga = (unsigned long long)(const void*)gsrc;
    u32x4 g0;
    g0[0] = 1u;                                        // count=1, user mode
    g0[1] = (unsigned)(unsigned long long)(void*)bpanel;  // lds_addr (bytes)
    g0[2] = (unsigned)ga;                              // global_addr[31:0]
    g0[3] = (unsigned)((ga >> 32) & 0x1ffffffu)        // global_addr[56:32]
            | 0x80000000u;                             // type=2 ("image")
    i32x8 g1;
    g1[0] = (int)0x00010000u;            // data_size=1 (2 bytes), no mask/pad
    g1[1] = (int)(768u << 16);           // tensor_dim0[15:0] = 768
    g1[2] = (int)(64u << 16);            // dim0 hi=0 | tensor_dim1[15:0] = 64
    g1[3] = (int)(768u << 16);           // dim1 hi=0 | tile_dim0 = 768
    g1[4] = 64;                          // tile_dim1 = 64, tile_dim2 = 0
    g1[5] = 768;                         // tensor_dim0_stride[31:0]
    g1[6] = 0;                           // stride hi | dim1_stride lo
    g1[7] = 0;
    i32x4 z4 = {0, 0, 0, 0};
    i32x8 z8 = {0, 0, 0, 0, 0, 0, 0, 0};
    __builtin_amdgcn_tensor_load_to_lds(g0, g1, z4, z4, z8, 0);
    __builtin_amdgcn_s_wait_tensorcnt(0);
  }
#else
  const uint4* s = (const uint4*)gsrc;
  uint4* d = (uint4*)bpanel;
  for (int i = tid; i < PANEL_SHORTS / 8; i += 256) d[i] = s[i];
#endif
  __syncthreads();
}

// ---------- f32 -> bf16 convert (vectorized x4) ----------
__global__ void cvt_f32_bf16_x4(const float4* __restrict__ src,
                                ushort4* __restrict__ dst, int n4) {
  int i = blockIdx.x * blockDim.x + threadIdx.x;
  if (i < n4) {
    float4 f = src[i];
    ushort4 o;
    o.x = f2bf(f.x); o.y = f2bf(f.y); o.z = f2bf(f.z); o.w = f2bf(f.w);
    dst[i] = o;
  }
}

// ---------- QKV GEMM: qkv = x @ Wqkv^T + b, scattered to Q (scaled), K, V^T ----------
// per-wave 32(M) x 64(N) tile; weight panel TDM-staged in LDS; A double-buffered
__global__ __launch_bounds__(256)
void qkv_gemm(const unsigned short* __restrict__ xbf,     // [MTOT, EMB] bf16
              const unsigned short* __restrict__ wbf,     // [N_QKV, EMB] bf16
              const float* __restrict__ bias,             // [N_QKV]
              unsigned short* __restrict__ Qb,            // [BH, SEQ, HD] (pre-scaled)
              unsigned short* __restrict__ Kb,            // [BH, SEQ, HD]
              unsigned short* __restrict__ Vt) {          // [BH, HD, SEQ]
  extern __shared__ unsigned short bpanel[];              // 64 x 768 bf16 (96 KB)
  const int lane = threadIdx.x & 31;
  const int wv   = threadIdx.x >> 5;
  const int half = lane >> 4;
  const int ln16 = lane & 15;

  const int nt = blockIdx.x % (N_QKV / 64);
  const int mt = blockIdx.x / (N_QKV / 64);
  const int m0 = mt * 256 + wv * 32;
  const int n0 = nt * 64;

  stage_panel_lds(bpanel, wbf + (size_t)n0 * EMB, threadIdx.x);

  const unsigned short* a0 = xbf + (m0 + ln16) * EMB + half * 16;
  const unsigned short* a1 = a0 + 16 * EMB;
  const unsigned short* w0 = bpanel + ln16 * EMB + half * 16;  // LDS panel

  f32x8 acc[2][4] = {};
  FragB a[2][2], b[2][4];

  ldfragA(a[0][0], a0);
  ldfragA(a[0][1], a1);
#pragma unroll
  for (int j = 0; j < 4; ++j) ldfragB(b[0][j], w0 + j * 16 * EMB);

#pragma unroll
  for (int it = 0; it < EMB / 32; ++it) {
    const int cur = it & 1, nxt = cur ^ 1;
    if (it + 1 < EMB / 32) {
      const int k = (it + 1) * 32;
      ldfragA(a[nxt][0], a0 + k);
      ldfragA(a[nxt][1], a1 + k);
#pragma unroll
      for (int j = 0; j < 4; ++j) ldfragB(b[nxt][j], w0 + j * 16 * EMB + k);
    }
#pragma unroll
    for (int j = 0; j < 4; ++j) {
      acc[0][j] = wmma_bf16(a[cur][0], b[cur][j], acc[0][j]);
      acc[1][j] = wmma_bf16(a[cur][1], b[cur][j], acc[1][j]);
    }
  }

  // epilogue: one (head, part) per 64-wide N tile since 192 = 3*64
  const int h    = n0 / 192;
  const int part = (n0 % 192) / 64;          // 0=Q 1=K 2=V
#pragma unroll
  for (int mi = 0; mi < 2; ++mi) {
#pragma unroll
    for (int j = 0; j < 4; ++j) {
      const int n  = n0 + 16 * j + ln16;
      const float bv = bias[n];
      const int d  = (16 * j + ln16) & 63;   // n % 64
#pragma unroll
      for (int r = 0; r < 8; ++r) {
        const int m  = m0 + 16 * mi + half * 8 + r;
        const int bb = m >> 11;              // / SEQ
        const int s  = m & (SEQ - 1);
        const float val = acc[mi][j][r] + bv;
        if (part == 0)
          Qb[(((bb * NH + h) * SEQ) + s) * HD + d] = f2bf(val * 0.125f); // 1/sqrt(64)
        else if (part == 1)
          Kb[(((bb * NH + h) * SEQ) + s) * HD + d] = f2bf(val);
        else
          Vt[(((bb * NH + h) * HD) + d) * SEQ + s] = f2bf(val);
      }
    }
  }
}

// ---------- flash attention: per wave 16 query rows, key tiles of 32, pipelined ----------
__global__ __launch_bounds__(256)
void attn(const unsigned short* __restrict__ Qb,
          const unsigned short* __restrict__ Kb,
          const unsigned short* __restrict__ Vt,
          unsigned short* __restrict__ ctx) {            // [NB, SEQ, EMB] bf16
  __shared__ unsigned short lds[8][16 * 32];             // per-wave P staging (1KB each)
  const int lane = threadIdx.x & 31;
  const int wv   = threadIdx.x >> 5;
  const int half = lane >> 4;
  const int ln16 = lane & 15;

  const int nqt = SEQ / 128;                             // 16
  const int bh  = blockIdx.x / nqt;
  const int q0  = (blockIdx.x % nqt) * 128 + wv * 16;

  const unsigned short* Qp = Qb + (size_t)bh * SEQ * HD;
  const unsigned short* Kp = Kb + (size_t)bh * SEQ * HD;
  const unsigned short* Vp = Vt + (size_t)bh * HD * SEQ;
  unsigned short* myLds = &lds[wv][0];

  // Q fragments (16x64 -> two 16x32 A-frags), loaded once
  FragB qf[2];
  {
    const unsigned short* p = Qp + (q0 + ln16) * HD + half * 16;
    ldfragA(qf[0], p);
    ldfragA(qf[1], p + 32);
  }

  f32x8 o[4] = {};
  float mrow[8], lrow[8];
#pragma unroll
  for (int r = 0; r < 8; ++r) { mrow[r] = -1e30f; lrow[r] = 0.0f; }

  auto loadK = [&](FragB (&kf)[2][2], int t0) {
#pragma unroll
    for (int g = 0; g < 2; ++g) {
      const unsigned short* p = Kp + (t0 + 16 * g + ln16) * HD + half * 16;
      ldfragB(kf[0][g], p);        // d = 0..31
      ldfragB(kf[1][g], p + 32);   // d = 32..63
    }
  };

  // one 32-key tile; issues V loads + next-K loads before the softmax VALU work
  auto tile = [&](int t0, FragB (&kcur)[2][2], FragB (&knext)[2][2], bool last) {
    FragB vf[4];
#pragma unroll
    for (int j = 0; j < 4; ++j)
      ldfragB(vf[j], Vp + (16 * j + ln16) * SEQ + t0 + half * 16);
    if (!last) loadK(knext, t0 + 32);
    if (t0 + 256 < SEQ) {  // pull 8 tiles ahead into cache (gfx1250 global_prefetch_b8)
      __builtin_prefetch(Kp + (t0 + 256 + lane) * HD, 0, 1);
      __builtin_prefetch(Vp + lane * SEQ + t0 + 256, 0, 1);
      __builtin_prefetch(Vp + (lane + 32) * SEQ + t0 + 256, 0, 1);
    }

    f32x8 sc[2] = {};
#pragma unroll
    for (int kk = 0; kk < 2; ++kk) {
      sc[0] = wmma_bf16(qf[kk], kcur[kk][0], sc[0]);
      sc[1] = wmma_bf16(qf[kk], kcur[kk][1], sc[1]);
    }

    // online softmax (row stats per (half, r); reduce across the 16-lane half)
#pragma unroll
    for (int r = 0; r < 8; ++r) {
      float tm = fmaxf(sc[0][r], sc[1][r]);
#pragma unroll
      for (int msk = 1; msk < 16; msk <<= 1) tm = fmaxf(tm, __shfl_xor(tm, msk, 32));
      const float mn    = fmaxf(mrow[r], tm);
      const float alpha = __expf(mrow[r] - mn);
      const float p0 = __expf(sc[0][r] - mn);
      const float p1 = __expf(sc[1][r] - mn);
      sc[0][r] = p0; sc[1][r] = p1;
      float ts = p0 + p1;
#pragma unroll
      for (int msk = 1; msk < 16; msk <<= 1) ts += __shfl_xor(ts, msk, 32);
      lrow[r] = lrow[r] * alpha + ts;
      mrow[r] = mn;
      o[0][r] *= alpha; o[1][r] *= alpha; o[2][r] *= alpha; o[3][r] *= alpha;
    }

    // P: C-layout f32 -> bf16 -> LDS (row-major 16x32) -> A-frag
#pragma unroll
    for (int r = 0; r < 8; ++r) {
      const int row = half * 8 + r;
      myLds[row * 32 + ln16]      = f2bf(sc[0][r]);
      myLds[row * 32 + 16 + ln16] = f2bf(sc[1][r]);
    }
    asm volatile("s_wait_dscnt 0" ::: "memory");   // per-wave LDS RAW, in-order DS
    FragB pf;
    ldfragA(pf, myLds + ln16 * 32 + half * 16);    // ds_load_b128 x2

    // PV: B-operand from V^T so K-pairs (t, t+1) are contiguous
#pragma unroll
    for (int j = 0; j < 4; ++j) o[j] = wmma_bf16(pf, vf[j], o[j]);
  };

  FragB kA[2][2], kB[2][2];
  loadK(kA, 0);
  for (int t = 0; t < SEQ; t += 64) {      // two tiles per iteration, explicit parity
    tile(t,      kA, kB, false);
    tile(t + 32, kB, kA, t + 64 >= SEQ);
  }

  // normalize + store ctx (bf16, [B,S,H*64])
  const int b = bh / NH, h = bh % NH;
#pragma unroll
  for (int r = 0; r < 8; ++r) {
    const int s = q0 + half * 8 + r;
    const float inv = 1.0f / lrow[r];
#pragma unroll
    for (int j = 0; j < 4; ++j) {
      const int d = 16 * j + ln16;
      ctx[((size_t)(b * SEQ + s)) * EMB + h * HD + d] = f2bf(o[j][r] * inv);
    }
  }
}

// ---------- output projection: out = ctx @ Wout^T + b_out (f32 out) ----------
__global__ __launch_bounds__(256)
void out_gemm(const unsigned short* __restrict__ abf,    // [MTOT, EMB] bf16
              const unsigned short* __restrict__ wbf,    // [EMB, EMB] bf16
              const float* __restrict__ bias,            // [EMB]
              float* __restrict__ out) {                 // [MTOT, EMB]
  extern __shared__ unsigned short bpanel[];             // 64 x 768 bf16 (96 KB)
  const int lane = threadIdx.x & 31;
  const int wv   = threadIdx.x >> 5;
  const int half = lane >> 4;
  const int ln16 = lane & 15;

  const int nt = blockIdx.x % (EMB / 64);
  const int mt = blockIdx.x / (EMB / 64);
  const int m0 = mt * 256 + wv * 32;
  const int n0 = nt * 64;

  stage_panel_lds(bpanel, wbf + (size_t)n0 * EMB, threadIdx.x);

  const unsigned short* a0 = abf + (m0 + ln16) * EMB + half * 16;
  const unsigned short* a1 = a0 + 16 * EMB;
  const unsigned short* w0 = bpanel + ln16 * EMB + half * 16;  // LDS panel

  f32x8 acc[2][4] = {};
  FragB a[2][2], b[2][4];

  ldfragA(a[0][0], a0);
  ldfragA(a[0][1], a1);
#pragma unroll
  for (int j = 0; j < 4; ++j) ldfragB(b[0][j], w0 + j * 16 * EMB);

#pragma unroll
  for (int it = 0; it < EMB / 32; ++it) {
    const int cur = it & 1, nxt = cur ^ 1;
    if (it + 1 < EMB / 32) {
      const int k = (it + 1) * 32;
      ldfragA(a[nxt][0], a0 + k);
      ldfragA(a[nxt][1], a1 + k);
#pragma unroll
      for (int j = 0; j < 4; ++j) ldfragB(b[nxt][j], w0 + j * 16 * EMB + k);
    }
#pragma unroll
    for (int j = 0; j < 4; ++j) {
      acc[0][j] = wmma_bf16(a[cur][0], b[cur][j], acc[0][j]);
      acc[1][j] = wmma_bf16(a[cur][1], b[cur][j], acc[1][j]);
    }
  }

#pragma unroll
  for (int mi = 0; mi < 2; ++mi) {
#pragma unroll
    for (int j = 0; j < 4; ++j) {
      const int n = n0 + 16 * j + ln16;
      const float bv = bias[n];
#pragma unroll
      for (int r = 0; r < 8; ++r) {
        const int m = m0 + 16 * mi + half * 8 + r;
        out[(size_t)m * EMB + n] = acc[mi][j][r] + bv;
      }
    }
  }
}

// ---------- launcher ----------
extern "C" void kernel_launch(void* const* d_in, const int* in_sizes, int n_in,
                              void* d_out, int out_size, void* d_ws, size_t ws_size,
                              hipStream_t stream) {
  (void)in_sizes; (void)n_in; (void)out_size; (void)ws_size;
  const float* x    = (const float*)d_in[0];
  const float* Wq   = (const float*)d_in[1];
  const float* bq   = (const float*)d_in[2];
  const float* Wo   = (const float*)d_in[3];
  const float* bo   = (const float*)d_in[4];
  float* out = (float*)d_out;

  const size_t NX  = (size_t)MTOT * EMB;       // 6.29M
  const size_t NWQ = (size_t)N_QKV * EMB;      // 1.77M
  const size_t NWO = (size_t)EMB * EMB;        // 0.59M
  const size_t NHD = (size_t)BH * SEQ * HD;    // 6.29M

  unsigned short* xbf = (unsigned short*)d_ws;
  unsigned short* wqb = xbf + NX;
  unsigned short* wob = wqb + NWQ;
  unsigned short* Qb  = wob + NWO;
  unsigned short* Kb  = Qb + NHD;
  unsigned short* Vt  = Kb + NHD;
  unsigned short* ctx = Vt + NHD;

  const int panelBytes = PANEL_SHORTS * 2;     // 96 KB dynamic LDS
  static bool attrDone = false;
  if (!attrDone) {   // host-side attribute set (not a stream op; capture-safe)
    (void)hipFuncSetAttribute((const void*)qkv_gemm,
                              hipFuncAttributeMaxDynamicSharedMemorySize, panelBytes);
    (void)hipFuncSetAttribute((const void*)out_gemm,
                              hipFuncAttributeMaxDynamicSharedMemorySize, panelBytes);
    attrDone = true;
  }

  cvt_f32_bf16_x4<<<(int)(NX  / 4 / 256), 256, 0, stream>>>((const float4*)x,  (ushort4*)xbf, (int)(NX  / 4));
  cvt_f32_bf16_x4<<<(int)(NWQ / 4 / 256), 256, 0, stream>>>((const float4*)Wq, (ushort4*)wqb, (int)(NWQ / 4));
  cvt_f32_bf16_x4<<<(int)(NWO / 4 / 256), 256, 0, stream>>>((const float4*)Wo, (ushort4*)wob, (int)(NWO / 4));

  qkv_gemm<<<(MTOT / 256) * (N_QKV / 64), 256, panelBytes, stream>>>(xbf, wqb, bq, Qb, Kb, Vt);
  attn    <<<BH * (SEQ / 128),            256, 0,          stream>>>(Qb, Kb, Vt, ctx);
  out_gemm<<<(MTOT / 256) * (EMB / 64),   256, panelBytes, stream>>>(ctx, wob, bo, out);
}
